// IMDCTLayer_36318243455293
// MI455X (gfx1250) — compile-verified, matrix operand order checked by reading, not run on hardware
//
#include <hip/hip_runtime.h>

typedef __attribute__((ext_vector_type(16))) _Float16 v16h_t;
typedef __attribute__((ext_vector_type(8)))  _Float16 v8h_t;
typedef __attribute__((ext_vector_type(8)))  float    v8f_t;
typedef __attribute__((ext_vector_type(4)))  float    v4f_t;
typedef __attribute__((ext_vector_type(4)))  unsigned int u32x4_t;
typedef __attribute__((ext_vector_type(8)))  int      i32x8_t;
typedef __attribute__((ext_vector_type(4)))  int      i32x4_t;

#define NWIN   576          // MDCT bins per frame (K of GEMM)
#define N2     1152         // frame length = 2*NWIN (N of GEMM)
#define HOP    288
#define NFRM   2048         // frames per batch
#define NB     16           // batch
#define OUTLEN 590688       // (NFRM-1)*HOP + N2, last frame ends exactly here
#define MROWS  (NB * NFRM)  // 32768 GEMM rows
#define KSTEPS (NWIN / 32)  // 18
#define CPW    144          // columns per wave (8 waves cover 1152)
#define SUBT   (CPW / 16)   // 9 sub-tiles of 16 cols per wave

// TDM pads 4 dwords after every 64 dwords stored -> staged LDS row pitch:
// 576 + 9*4 = 612 dwords = 2448 B. 2448/16 = 153 = 9 mod 16 (coprime with 16)
// => 16 lanes reading 16 consecutive rows via ds_load_b128 are conflict-free.
#define LDS_ROW_DW 612
#define LDS_FLOATS (16 * LDS_ROW_DW)   // 9792 floats = 39168 B

// padded LDS byte offset of f32 element k within a staged row
__device__ __forceinline__ int pad_off(int k) {
    return 4 * k + ((k >> 6) << 4);    // +16 B after each 64-dword chunk
}

// ---------------------------------------------------------------------------
// Kernel 1: build folded constant matrix Bt[n2][k] (f16, K-contiguous rows).
// Bt[n][k] = w2n[n] * sgn * (1/N) * cos((pi/N)*(k*(2m+1)+0.5))
//   n <  N : m = N-1-n, sgn = -1   (the "-reverse(yr)" half)
//   n >= N : m = n-N,   sgn = +1
// ---------------------------------------------------------------------------
__global__ void build_Bt(_Float16* __restrict__ Bt) {
    int idx = blockIdx.x * blockDim.x + threadIdx.x;
    if (idx >= N2 * NWIN) return;
    int n = idx / NWIN;
    int k = idx % NWIN;
    int m; float sgn;
    if (n < NWIN) { m = NWIN - 1 - n; sgn = -1.0f; }
    else          { m = n - NWIN;     sgn =  1.0f; }
    int p = (k * (2 * m + 1)) % (2 * NWIN);            // max 575*1151 fits int32
    float c = __cosf(((float)p + 0.5f) * (3.14159265358979323846f / (float)NWIN));
    float w = __sinf(((float)n + 0.5f) * (3.14159265358979323846f / (float)N2));
    Bt[idx] = (_Float16)(sgn * w * c * (1.0f / (float)NWIN));
}

// ---------------------------------------------------------------------------
// Kernel 2: zero the OLA output (harness poisons d_out).
// ---------------------------------------------------------------------------
__global__ void zero_out(float* __restrict__ out, int n) {
    int i = blockIdx.x * blockDim.x + threadIdx.x;
    if (i < n) out[i] = 0.0f;
}

// ---------------------------------------------------------------------------
// Kernel 3: frames = X @ B via V_WMMA_F32_16X16X32_F16, fused overlap-add.
// One block = one 16-row tile of X, staged to LDS once by the Tensor Data
// Mover (2D tile 16x576 f32, padded pitch for conflict-free ds_load_b128),
// then 8 waves each compute 16 rows x 144 cols (9 f32 accum fragments).
// ---------------------------------------------------------------------------
__global__ __launch_bounds__(256) void imdct_gemm(
        const float*    __restrict__ X,     // [MROWS][NWIN] f32
        const _Float16* __restrict__ Bt,    // [N2][NWIN] f16
        float*          __restrict__ out) { // [NB][OUTLEN] f32
    __shared__ float A_lds[LDS_FLOATS];     // single LDS object -> offset 0

    const int tileM = blockIdx.x;           // 0..2047
    const int w     = threadIdx.x >> 5;     // wave in block, 0..7
    const int lane  = threadIdx.x & 31;
    const int g     = lane >> 4;
    const int c     = lane & 15;

    // ---- TDM: stage A tile (16 x 576 f32) global -> LDS, wave 0 issues ----
    if (threadIdx.x < 32) {
        unsigned long long ga =
            (unsigned long long)(uintptr_t)(X + (size_t)tileM * 16 * NWIN);
        u32x4_t g0;
        g0[0] = 1u;                                   // count=1, gather off
        g0[1] = 0u;                                   // lds_addr = 0
        g0[2] = (unsigned)(ga & 0xffffffffull);       // global_addr[31:0]
        g0[3] = (unsigned)((ga >> 32) & 0x01ffffffull)
              | (2u << 30);                           // global_addr[56:32], type=2
        i32x8_t g1;
        g1[0] = (int)((2u << 16)      // data_size = 4 B
                    | (1u << 20)      // pad_enable
                    | (5u << 22)      // pad_interval: 64 dwords
                    | (3u << 25));    // pad_amount:   4 dwords
        g1[1] = (int)((unsigned)NWIN << 16);    // tensor_dim0[15:0]=576
        g1[2] = (int)((unsigned)MROWS << 16);   // dim0 hi=0 | tensor_dim1 lo=32768
        g1[3] = (int)((unsigned)NWIN << 16);    // dim1 hi=0 | tile_dim0=576
        g1[4] = 16;                             // tile_dim1=16 | tile_dim2=0
        g1[5] = NWIN;                           // tensor_dim0_stride = 576
        g1[6] = 0;
        g1[7] = 0;
        i32x4_t gz4 = {0, 0, 0, 0};             // 2D: groups 2/3 unused
        i32x8_t gz8 = {0, 0, 0, 0, 0, 0, 0, 0};
        __builtin_amdgcn_tensor_load_to_lds(g0, g1, gz4, gz4, gz8, 0);
        __builtin_amdgcn_s_wait_tensorcnt((short)0);
    }
    __syncthreads();

    // The compiler cannot see the TDM's LDS write (A_lds's address only went
    // into the descriptor as an integer). Escape the pointer into an asm with
    // a memory clobber so the A_lds loads below are not folded to undef.
    asm volatile("" : : "r"(&A_lds[0]) : "memory");

    // ---- GEMM: 16 rows x 144 cols per wave -----------------------------
    const char*     lrow = (const char*)A_lds + c * (LDS_ROW_DW * 4);
    const _Float16* Bw   = Bt + (size_t)(w * CPW + c) * NWIN;

    v8f_t acc[SUBT];
    #pragma unroll
    for (int s = 0; s < SUBT; ++s) acc[s] = (v8f_t){};

    for (int kt = 0; kt < KSTEPS; ++kt) {
        const int k0 = kt * 32;
        const int ka = k0 + g * 8;              // first 8-dword K-run
        const int kb = ka + 16;                 // second 8-dword K-run
        const int pa = pad_off(ka);
        const int pb = pad_off(kb);

        // A fragment from LDS (f32 -> f16 cvt), reused across 9 WMMAs
        v4f_t af0 = *(const v4f_t*)(lrow + pa);
        v4f_t af1 = *(const v4f_t*)(lrow + pa + 16);
        v4f_t af2 = *(const v4f_t*)(lrow + pb);
        v4f_t af3 = *(const v4f_t*)(lrow + pb + 16);
        v16h_t a;
        #pragma unroll
        for (int t = 0; t < 4; ++t) {
            a[t]      = (_Float16)af0[t];
            a[4 + t]  = (_Float16)af1[t];
            a[8 + t]  = (_Float16)af2[t];
            a[12 + t] = (_Float16)af3[t];
        }

        // 9 column sub-tiles: B fragment = 2 x 128-bit f16 loads (L2-resident)
        #pragma unroll
        for (int s = 0; s < SUBT; ++s) {
            const _Float16* bp = Bw + (size_t)s * 16 * NWIN + ka;
            v8h_t b0 = *(const v8h_t*)(bp);
            v8h_t b1 = *(const v8h_t*)(bp + 16);
            v16h_t b;
            #pragma unroll
            for (int t = 0; t < 8; ++t) { b[t] = b0[t]; b[8 + t] = b1[t]; }

            acc[s] = __builtin_amdgcn_wmma_f32_16x16x32_f16(
                /*neg_a=*/false, a, /*neg_b=*/false, b,
                /*c_mod=*/(short)0, acc[s],
                /*reuse_a=*/false, /*reuse_b=*/false);
        }
    }

    // ---- fused overlap-add epilogue ------------------------------------
    // acc[s][i]: GEMM row = tileM*16 + g*8 + i, col n2 = w*144 + s*16 + c.
    const int row0 = tileM * 16 + g * 8;
    #pragma unroll
    for (int s = 0; s < SUBT; ++s) {
        const int n2 = w * CPW + s * 16 + c;
        #pragma unroll
        for (int i = 0; i < 8; ++i) {
            const int frame = row0 + i;          // flat (b, f)
            const int bidx  = frame >> 11;       // / NFRM
            const int f     = frame & (NFRM - 1);
            atomicAdd(out + (size_t)bidx * OUTLEN + (size_t)f * HOP + n2,
                      acc[s][i]);
        }
    }
}

// ---------------------------------------------------------------------------
extern "C" void kernel_launch(void* const* d_in, const int* in_sizes, int n_in,
                              void* d_out, int out_size, void* d_ws, size_t ws_size,
                              hipStream_t stream) {
    const float* X   = (const float*)d_in[0];
    float*       out = (float*)d_out;
    _Float16*    Bt  = (_Float16*)d_ws;          // 1152*576*2 B = 1.27 MB

    // 1) constant matrix
    {
        int n = N2 * NWIN;
        build_Bt<<<(n + 255) / 256, 256, 0, stream>>>(Bt);
    }
    // 2) zero output for atomic overlap-add
    {
        zero_out<<<(out_size + 255) / 256, 256, 0, stream>>>(out, out_size);
    }
    // 3) TDM-staged WMMA GEMM + fused OLA: one block per 16-row tile
    {
        imdct_gemm<<<MROWS / 16, 256, 0, stream>>>(X, Bt, out);
    }
}
